// EncoderRNN_54065048322518
// MI455X (gfx1250) — matrix operand order repaired and checked by reading, not compiled, not run on hardware
//
#include <hip/hip_runtime.h>

// ---------------------------------------------------------------------------
// EncoderRNN (embedding + LSTM) for MI455X / gfx1250, wave32 + WMMA f16 + TDM.
//   VOCAB=32000 EMBED=512 HIDDEN=1024 B=32 T=512
// Plan:
//   1) prep: embedding gather -> f16 [T][B][E]; cast W_ih/W_hh/h0 -> f16; zero flags
//   2) GEMM x_proj = emb @ W_ih^T: wave tile 32x64, B tile staged in LDS by the
//      Tensor Data Mover (tensor_load_to_lds, double buffered, s_wait_tensorcnt),
//      ds_load fragments, v_wmma_f32_16x16x32_f16.
//   3) ONE persistent kernel for all 512 LSTM steps (16 co-resident blocks),
//      step hand-off via agent-scope release/acquire flags; c kept in registers;
//      W_hh (8MB f16) and h ping-pong bufs stay resident in the 192MB L2.
// ---------------------------------------------------------------------------

typedef _Float16 v16h __attribute__((ext_vector_type(16)));
typedef _Float16 v8h  __attribute__((ext_vector_type(8)));
typedef _Float16 v4h  __attribute__((ext_vector_type(4)));
typedef float    v8f  __attribute__((ext_vector_type(8)));
typedef unsigned int uint32x4 __attribute__((ext_vector_type(4)));
typedef int          int32x4  __attribute__((ext_vector_type(4)));
typedef int          int32x8  __attribute__((ext_vector_type(8)));

#define EMBED  512
#define HIDDEN 1024
#define GATES  4096   // 4*HIDDEN
#define BATCH  32
#define SEQ    512

// ---- WMMA fragment loaders (layouts per CDNA5 ISA 7.12.2, wave32) ----------

// A (16-bit, 16x32, row-major, leading dim `ld` halfs).
// lane L: row M=L%16; halfs 0..7 = K (L<16?0:8)+0..7 ; halfs 8..15 = +16.
__device__ __forceinline__ v16h load_a_frag(const _Float16* __restrict__ base, int ld) {
    const int lane = threadIdx.x & 31;
    const _Float16* p = base + (lane & 15) * ld + ((lane < 16) ? 0 : 8);
    v8h lo = *(const v8h*)(p);
    v8h hi = *(const v8h*)(p + 16);
    return __builtin_shufflevector(lo, hi, 0,1,2,3,4,5,6,7,8,9,10,11,12,13,14,15);
}

// B (16-bit, 32x16): B[k][n]=W[n][k]; lane L: col N=L%16; halfs 0..7 = K (L<16?0:8)+0..7 ;
// halfs 8..15 = +16. Row stride `ld` halfs (16B-aligned is enough).
__device__ __forceinline__ v16h load_b_frag(const _Float16* __restrict__ wrow0, int ld) {
    const int lane = threadIdx.x & 31;
    const _Float16* p = wrow0 + (lane & 15) * ld + ((lane < 16) ? 0 : 8);
    v8h lo = *(const v8h*)(p);
    v8h hi = *(const v8h*)(p + 16);
    return __builtin_shufflevector(lo, hi, 0,1,2,3,4,5,6,7,8,9,10,11,12,13,14,15);
}

__device__ __forceinline__ float sigmoidf_fast(float x) {
    return 1.0f / (1.0f + __expf(-x));
}

// ---- Tensor Data Mover: 2D tile (tile_h rows x 32 halfs) global->LDS -------
// Pads 16B after every 64B row => 80B LDS row stride (bank-spread ds reads).
// D# layout per CDNA5 ISA.
__device__ __forceinline__ void tdm_load_rows32h(
    unsigned lds_off, const _Float16* gsrc, int tile_h,
    int tensor_w, int tensor_h, int row_stride)
{
    unsigned long long ga = (unsigned long long)(size_t)gsrc;
    uint32x4 g0;
    g0[0] = 1u;                                        // count=1 (valid, user)
    g0[1] = lds_off;                                   // lds_addr
    g0[2] = (unsigned)(ga & 0xFFFFFFFFu);              // global_addr[31:0]
    g0[3] = (unsigned)((ga >> 32) & 0x01FFFFFFu) | (2u << 30);  // [56:32], type=2
    int32x8 g1;
    g1[0] = (1 << 16) | (1 << 20) | (3 << 22) | (3 << 25);  // 2B elems, pad: 16 dw interval, 4 dw amount
    g1[1] = (tensor_w & 0xFFFF) << 16;                 // tensor_dim0 lo16
    g1[2] = ((tensor_w >> 16) & 0xFFFF) | ((tensor_h & 0xFFFF) << 16);
    g1[3] = ((tensor_h >> 16) & 0xFFFF) | (32 << 16);  // tile_dim0 = 32 halfs
    g1[4] = tile_h & 0xFFFF;                           // tile_dim1
    g1[5] = row_stride;                                // tensor_dim0_stride lo32
    g1[6] = 0;
    g1[7] = 0;
    int32x4 z4 = {0, 0, 0, 0};
#if __clang_major__ >= 23
    int32x8 z8 = {0, 0, 0, 0, 0, 0, 0, 0};
    __builtin_amdgcn_tensor_load_to_lds(g0, g1, z4, z4, z8, 0);
#else
    __builtin_amdgcn_tensor_load_to_lds(g0, g1, z4, z4, 0);
#endif
}

// ---- prep: embedding gather + f32->f16, output [T][B][E] -------------------
__global__ __launch_bounds__(256) void embed_cast_kernel(
    const int* __restrict__ ids, const float* __restrict__ emb,
    _Float16* __restrict__ dst)
{
    int i = blockIdx.x * 256 + threadIdx.x;
    int tb = i >> 7;                      // E/4 = 128 float4 chunks per row
    int e4 = (i & 127) << 2;
    int t = tb >> 5, b = tb & 31;
    int id = ids[b * SEQ + t];
    float4 v = *(const float4*)(emb + id * EMBED + e4);
    v4h o = { (_Float16)v.x, (_Float16)v.y, (_Float16)v.z, (_Float16)v.w };
    *(v4h*)(dst + tb * EMBED + e4) = o;
}

__global__ __launch_bounds__(256) void cast_f16_kernel(
    const float* __restrict__ src, _Float16* __restrict__ dst, int n)
{
    int i = (blockIdx.x * 256 + threadIdx.x) << 2;
    if (i < n) {
        float4 v = *(const float4*)(src + i);
        v4h o = { (_Float16)v.x, (_Float16)v.y, (_Float16)v.z, (_Float16)v.w };
        *(v4h*)(dst + i) = o;
    }
}

__global__ __launch_bounds__(256) void zero_flags_kernel(int* __restrict__ flags, int n) {
    int i = blockIdx.x * 256 + threadIdx.x;
    if (i < n) flags[i] = 0;
}

// ---- GEMM: xproj[M=16384][N=4096] = A[M][512] @ W[N][512]^T ----------------
// 8 waves/block; wave tile 32M x 64N (block 256M x 64N). B tile (64 rows x 32
// halfs per K-step) TDM-staged into LDS (each wave DMAs its own 8-row slice),
// double-buffered with s_wait_tensorcnt; fragments read via ds_load.
#define BROW 40   // LDS halfs per B row (64B data + 16B TDM pad)

__global__ __launch_bounds__(256) void gemm_xproj_kernel(
    const _Float16* __restrict__ A,   // [16384][512]
    const _Float16* __restrict__ W,   // [4096][512]
    float* __restrict__ C)            // [16384][4096]
{
    constexpr int K = EMBED;
    __shared__ _Float16 bstage[2][64 * BROW];     // 2 x 5120B
    const int wave = threadIdx.x >> 5;
    const int lane = threadIdx.x & 31;
    const int m0   = (blockIdx.y * 8 + wave) * 32;
    const int n0   = blockIdx.x * 64;

    const _Float16* wsrc = W + (n0 + wave * 8) * K;     // this wave's 8-row slice
    const unsigned  lds0 = (unsigned)(size_t)&bstage[0][wave * 8 * BROW];
    const unsigned  lds1 = (unsigned)(size_t)&bstage[1][wave * 8 * BROW];

    tdm_load_rows32h(lds0, wsrc, 8, K, GATES, K);       // prologue: k=0 -> buf0

    v8f acc[2][4] = {};
    for (int kk = 0; kk < 16; ++kk) {
        const int k = kk * 32;
        if (kk < 15) {
            tdm_load_rows32h((kk & 1) ? lds0 : lds1, wsrc + k + 32, 8, K, GATES, K);
            __builtin_amdgcn_s_wait_tensorcnt((short)1);   // buf[kk&1] ready
        } else {
            __builtin_amdgcn_s_wait_tensorcnt((short)0);
        }
        __syncthreads();                                    // whole tile staged
        const _Float16* bb = &bstage[kk & 1][0];
        v16h a0 = load_a_frag(A + (size_t)m0 * K + k, K);
        v16h a1 = load_a_frag(A + (size_t)(m0 + 16) * K + k, K);
#pragma unroll
        for (int j = 0; j < 4; ++j) {
            v16h b = load_b_frag(bb + j * 16 * BROW, BROW); // ds_load fragments
            acc[0][j] = __builtin_amdgcn_wmma_f32_16x16x32_f16(
                false, a0, false, b, (short)0, acc[0][j], false, false);
            acc[1][j] = __builtin_amdgcn_wmma_f32_16x16x32_f16(
                false, a1, false, b, (short)0, acc[1][j], false, false);
        }
        __syncthreads();                                    // reads done before re-stage
    }

    const int mofs = (lane >> 4) * 8;
    const int ncol = lane & 15;
#pragma unroll
    for (int i = 0; i < 2; ++i)
#pragma unroll
        for (int j = 0; j < 4; ++j)
#pragma unroll
            for (int r = 0; r < 8; ++r)
                C[(size_t)(m0 + i * 16 + r + mofs) * GATES + n0 + j * 16 + ncol] = acc[i][j][r];
}

// ---- persistent LSTM: all 512 steps in one launch --------------------------
// 16 blocks x 8 waves. Wave: m0=(w&1)*16 batch rows, 16 hidden cols; computes
// gates i,f,g,o for its tile (elementwise fusion wave-local, c in registers).
// Cross-block step hand-off: release-atomic flag per step + acquire spin.
__global__ __launch_bounds__(256) void lstm_persistent_kernel(
    const float*    __restrict__ xproj,   // [T][B][4H]
    const _Float16* __restrict__ Whh,     // [4096][1024] f16
    const float*    __restrict__ c0,      // [B][H]
    _Float16*       __restrict__ hb0,     // h(t) lives in hb[t&1]; hb0 = h(0)
    _Float16*       __restrict__ hb1,
    int*            __restrict__ flags,   // [SEQ+1], zeroed
    float*          __restrict__ out)     // [B][T][H] then hn[B][H], cn[B][H]
{
    constexpr int K = HIDDEN;
    const int wave = threadIdx.x >> 5;
    const int lane = threadIdx.x & 31;
    const int m0   = (wave & 1) * 16;
    const int j0   = blockIdx.x * 64 + (wave >> 1) * 16;
    const int mofs = (lane >> 4) * 8;
    const int ncol = lane & 15;
    const int col  = j0 + ncol;

    float creg[8];
#pragma unroll
    for (int r = 0; r < 8; ++r)
        creg[r] = c0[(m0 + r + mofs) * HIDDEN + col];

    float* hnB = out + (size_t)BATCH * SEQ * HIDDEN;
    float* cnB = hnB + BATCH * HIDDEN;

    for (int t = 0; t < SEQ; ++t) {
        const _Float16* hrd = (t & 1) ? hb1 : hb0;
        _Float16*       hwr = (t & 1) ? hb0 : hb1;
        if (t > 0) {
            if (threadIdx.x == 0)
                while (__hip_atomic_load(&flags[t], __ATOMIC_ACQUIRE,
                                         __HIP_MEMORY_SCOPE_AGENT) < 16) {}
            __syncthreads();
        }
        v8f acc[4] = {};
        for (int k = 0; k < K; k += 32) {
            v16h a = load_a_frag(hrd + m0 * K + k, K);
            __builtin_prefetch(Whh + (size_t)j0 * K + k + 64, 0, 1);
#pragma unroll
            for (int g = 0; g < 4; ++g) {
                v16h b = load_b_frag(Whh + (size_t)(g * HIDDEN + j0) * K + k, K);
                acc[g] = __builtin_amdgcn_wmma_f32_16x16x32_f16(
                    false, a, false, b, (short)0, acc[g], false, false);
            }
        }
        const float* xp = xproj + (size_t)t * BATCH * GATES;
#pragma unroll
        for (int r = 0; r < 8; ++r) {
            const int brow = m0 + r + mofs;
            const float* x = xp + (size_t)brow * GATES + col;
            float gi = x[0 * HIDDEN] + acc[0][r];
            float gf = x[1 * HIDDEN] + acc[1][r];
            float gg = x[2 * HIDDEN] + acc[2][r];
            float go = x[3 * HIDDEN] + acc[3][r];
            float c_new = sigmoidf_fast(gf) * creg[r] + sigmoidf_fast(gi) * tanhf(gg);
            float h_new = sigmoidf_fast(go) * tanhf(c_new);
            creg[r] = c_new;
            hwr[brow * HIDDEN + col] = (_Float16)h_new;
            out[((size_t)brow * SEQ + t) * HIDDEN + col] = h_new;
            if (t == SEQ - 1) {
                hnB[brow * HIDDEN + col] = h_new;
                cnB[brow * HIDDEN + col] = c_new;
            }
        }
        __threadfence();                 // my h stores visible at device scope
        __syncthreads();                 // whole block done
        if (threadIdx.x == 0)
            __hip_atomic_fetch_add(&flags[t + 1], 1, __ATOMIC_RELEASE,
                                   __HIP_MEMORY_SCOPE_AGENT);
    }
}

// ---------------------------------------------------------------------------
extern "C" void kernel_launch(void* const* d_in, const int* in_sizes, int n_in,
                              void* d_out, int out_size, void* d_ws, size_t ws_size,
                              hipStream_t stream) {
    const int*   ids = (const int*)d_in[0];
    const float* h0  = (const float*)d_in[1];
    const float* c0  = (const float*)d_in[2];
    const float* emb = (const float*)d_in[3];
    const float* Wih = (const float*)d_in[4];
    const float* Whh = (const float*)d_in[5];
    float*       out = (float*)d_out;
    char*        ws  = (char*)d_ws;

    _Float16* embh  = (_Float16*)(ws);                        // 16 MB
    _Float16* wihh  = (_Float16*)(ws + (16u << 20));          // 4 MB
    _Float16* whhh  = (_Float16*)(ws + (20u << 20));          // 8 MB
    float*    xproj = (float*)   (ws + (28u << 20));          // 256 MB
    char*     tail  = ws + (28u << 20) + ((size_t)SEQ * BATCH * GATES * 4);
    _Float16* hbuf0 = (_Float16*)(tail);                      // 64 KB
    _Float16* hbuf1 = (_Float16*)(tail + (64u << 10));        // 64 KB
    int*      flags = (int*)     (tail + (128u << 10));       // (SEQ+1) ints

    // 1) prep
    embed_cast_kernel<<<(SEQ * BATCH * EMBED / 4) / 256, 256, 0, stream>>>(ids, emb, embh);
    cast_f16_kernel<<<(GATES * EMBED  / 4) / 256, 256, 0, stream>>>(Wih, wihh, GATES * EMBED);
    cast_f16_kernel<<<(GATES * HIDDEN / 4) / 256, 256, 0, stream>>>(Whh, whhh, GATES * HIDDEN);
    cast_f16_kernel<<<(BATCH * HIDDEN / 4) / 256, 256, 0, stream>>>(h0, hbuf0, BATCH * HIDDEN);
    zero_flags_kernel<<<3, 256, 0, stream>>>(flags, SEQ + 1);

    // 2) input-projection GEMM (TDM-staged B, WMMA f16 -> f32)
    gemm_xproj_kernel<<<dim3(GATES / 64, (SEQ * BATCH) / 256), 256, 0, stream>>>(embh, wihh, xproj);

    // 3) full recurrence in one persistent launch
    lstm_persistent_kernel<<<HIDDEN / 64, 256, 0, stream>>>(
        xproj, whhh, c0, hbuf0, hbuf1, flags, out);
}
